// MultiHeadAttention_79465484910864
// MI455X (gfx1250) — matrix operand order repaired and checked by reading, not compiled, not run on hardware
//
#include <hip/hip_runtime.h>

typedef __bf16 bf16;
typedef __attribute__((ext_vector_type(16))) __bf16 v16bf;
typedef __attribute__((ext_vector_type(8)))  float  v8f;
typedef __attribute__((ext_vector_type(4)))  unsigned int u32x4;
typedef __attribute__((ext_vector_type(4)))  int v4i;

#define B_ 2
#define T_ 2048
#define M_ 1024
#define H_ 8
#define D_ 128

union FragU { v16bf v; u32x4 u[2]; };

__device__ __forceinline__ v16bf load_frag(const bf16* p0, const bf16* p1) {
  FragU f;
  f.u[0] = *(const u32x4*)p0;
  f.u[1] = *(const u32x4*)p1;
  return f.v;
}

__device__ __forceinline__ v8f wmma_bf16(v16bf a, v16bf b, v8f c) {
  return __builtin_amdgcn_wmma_f32_16x16x32_bf16(false, a, false, b, (short)0, c, false, false);
}

// ---- CDNA5 async global->LDS copy (ASYNCcnt), with safe fallback ----------
#if defined(__AMDGCN__) && __has_builtin(__builtin_amdgcn_global_load_async_to_lds_b128)
#define HAVE_ASYNC_LDS 1
#else
#define HAVE_ASYNC_LDS 0
#endif

__device__ __forceinline__ void async_b128(const bf16* g, bf16* l) {
#if HAVE_ASYNC_LDS
  __builtin_amdgcn_global_load_async_to_lds_b128(
      (__attribute__((address_space(1))) v4i*)g,
      (__attribute__((address_space(3))) v4i*)l, 0, 0);
#else
  *(u32x4*)l = *(const u32x4*)g;     // sync bounce through VGPRs
#endif
}

template <int N>
__device__ __forceinline__ void wait_async() {
#if HAVE_ASYNC_LDS
#if __has_builtin(__builtin_amdgcn_s_wait_asynccnt)
  __builtin_amdgcn_s_wait_asynccnt(N);
#else
  asm volatile("s_wait_asynccnt %0" :: "i"(N) : "memory");
#endif
#endif
}

// ---------------------------------------------------------------------------
// Kernel 0: fp32 -> bf16 packing + weight transposes.
//   xb  [B*T, M];  wt [H][384][1024] (col-major per head, q|k|v);  wot [1024][1024]
// ---------------------------------------------------------------------------
__global__ __launch_bounds__(256) void k_convert(const float* __restrict__ x,
                                                 const float* __restrict__ wq,
                                                 const float* __restrict__ wk,
                                                 const float* __restrict__ wvv,
                                                 const float* __restrict__ wo,
                                                 bf16* __restrict__ xb,
                                                 bf16* __restrict__ wt,
                                                 bf16* __restrict__ wot) {
  long i = (long)blockIdx.x * 256 + threadIdx.x;
  const long NX = (long)B_ * T_ * M_;
  const long NW = (long)H_ * 384 * 1024;
  const long NO = (long)1024 * 1024;
  if (i < NX) { xb[i] = (bf16)x[i]; return; }
  i -= NX;
  if (i < NW) {
    long h = i / (384 * 1024);
    long rem = i - h * (384 * 1024);
    long c = rem >> 10;
    long k = rem & 1023;
    float val;
    if (c < 128)      val = wq [(k * H_ + h) * D_ + c];
    else if (c < 256) val = wk [(k * H_ + h) * D_ + (c - 128)];
    else              val = wvv[(k * H_ + h) * D_ + (c - 256)];
    wt[h * (384 * 1024) + c * 1024 + k] = (bf16)val;
    return;
  }
  i -= NW;
  if (i < NO) {
    long c = i >> 10, k = i & 1023;
    wot[i] = (bf16)wo[k * 1024 + c];
  }
}

// ---------------------------------------------------------------------------
// Kernel 1: fused QKV projection, async double-buffered LDS staging.
// Grid (B*T/64, H, 2): 64-row slab x 192-col half-panel of [q|k|v].
// 8 waves: (row-tile rt 0..3, col-half 0..1) -> 6 accumulators each.
// ---------------------------------------------------------------------------
__global__ __launch_bounds__(256) void k_qkv(const bf16* __restrict__ xb,
                                             const bf16* __restrict__ wt,
                                             bf16* __restrict__ Qr,
                                             bf16* __restrict__ Kr,
                                             bf16* __restrict__ Vt) {
  __shared__ __align__(16) bf16 sW[2][192 * 32];   // 12 KB per buffer
  __shared__ __align__(16) bf16 sX[2][64 * 32];    //  4 KB per buffer
  const int tid  = threadIdx.x;
  const int lane = tid & 31;
  const int wv   = tid >> 5;
  const int hi   = lane >> 4;
  const int ln   = lane & 15;
  const int rt   = wv & 3;
  const int half = wv >> 2;
  const int rowblk = blockIdx.x * 64;
  const int h  = blockIdx.y;
  const int cb = blockIdx.z;                        // which 192-col half
  const bf16* wth   = wt + (size_t)h * (384 * 1024) + (size_t)cb * (192 * 1024);
  const bf16* xrows = xb + (size_t)rowblk * M_;

  auto stage = [&](int k0, int bsel) {
    // W: 192 rows x 32 k = 768 x 16B chunks -> 3/thread, coalesced
#pragma unroll
    for (int i = 0; i < 3; ++i) {
      int c = i * 256 + tid;
      int row = c >> 2, off = (c & 3) * 8;
      async_b128(wth + (size_t)row * 1024 + k0 + off, &sW[bsel][row * 32 + off]);
    }
    // x: 256 x 16B chunks -> 1/thread
    {
      int row = tid >> 2, off = (tid & 3) * 8;
      async_b128(xrows + (size_t)row * M_ + k0 + off, &sX[bsel][row * 32 + off]);
    }
  };

  v8f acc[6] = {};
  stage(0, 0);
  for (int kk = 0; kk < 32; ++kk) {
    int cur = kk & 1;
    if (kk + 1 < 32) { stage((kk + 1) * 32, cur ^ 1); wait_async<4>(); }
    else             { wait_async<0>(); }
    __syncthreads();
    const bf16* ap = &sX[cur][(rt * 16 + ln) * 32 + hi * 8];
    v16bf af = load_frag(ap, ap + 16);
    v16bf bfr[6];
#pragma unroll
    for (int j = 0; j < 6; ++j) {
      const bf16* bp = &sW[cur][((half * 6 + j) * 16 + ln) * 32 + hi * 8];
      bfr[j] = load_frag(bp, bp + 16);
    }
#pragma unroll
    for (int j = 0; j < 6; ++j) acc[j] = wmma_bf16(af, bfr[j], acc[j]);
    __syncthreads();
  }

  const int rowbase = rowblk + rt * 16;
#pragma unroll
  for (int j = 0; j < 6; ++j) {
    int gct = cb * 12 + half * 6 + j;               // global 16-col tile, 0..23
#pragma unroll
    for (int r = 0; r < 8; ++r) {
      int grow = rowbase + r + hi * 8;              // flattened b*T + t
      int gb = grow >> 11;
      int t  = grow & (T_ - 1);
      bf16 val = (bf16)acc[j][r];
      if (gct < 8) {
        Qr[(((size_t)gb * H_ + h) * T_ + t) * D_ + gct * 16 + ln] = val;
      } else if (gct < 16) {
        Kr[(((size_t)gb * H_ + h) * T_ + t) * D_ + (gct - 8) * 16 + ln] = val;
      } else {
        int d = (gct - 16) * 16 + ln;
        Vt[(((size_t)gb * H_ + h) * D_ + d) * T_ + t] = val;
      }
    }
  }
}

// ---------------------------------------------------------------------------
// Kernel 2: in-place RMSNorm + RoPE + sqrt(1/128) on Qr then Kr.
// ---------------------------------------------------------------------------
__global__ __launch_bounds__(256) void k_normrope(bf16* __restrict__ Qr,
                                                  bf16* __restrict__ Kr) {
  const int lane = threadIdx.x & 31;
  const int wv   = threadIdx.x >> 5;
  long row = (long)blockIdx.x * 8 + wv;
  const long NR = (long)B_ * H_ * T_;
  bf16* base = (row < NR) ? Qr : Kr;
  long r = (row < NR) ? row : row - NR;
  int t = (int)(r & (T_ - 1));
  bf16* p = base + r * D_ + lane * 4;

  float v[4];
#pragma unroll
  for (int s = 0; s < 4; ++s) v[s] = (float)p[s];
  float ss = v[0] * v[0] + v[1] * v[1] + v[2] * v[2] + v[3] * v[3];
  ss += __shfl_xor(ss, 1, 32);
  ss += __shfl_xor(ss, 2, 32);
  ss += __shfl_xor(ss, 4, 32);
  ss += __shfl_xor(ss, 8, 32);
  ss += __shfl_xor(ss, 16, 32);
  float scale = rsqrtf(ss * (1.0f / 128.0f) + 1e-6f) * 0.08838834764831845f;
#pragma unroll
  for (int s = 0; s < 4; ++s) v[s] *= scale;

  float out[4];
#pragma unroll
  for (int s = 0; s < 4; ++s) {
    float partner = __shfl_xor(v[s], 16, 32);
    int idx = (lane & 15) * 4 + s;
    float freq = __expf(-(float)idx * (9.210340371976184f / 64.0f));
    float sn, cs;
    __sincosf((float)t * freq, &sn, &cs);
    if (lane < 16) out[s] = v[s] * cs - partner * sn;
    else           out[s] = partner * sn + v[s] * cs;
  }
#pragma unroll
  for (int s = 0; s < 4; ++s) p[s] = (bf16)out[s];
}

// ---------------------------------------------------------------------------
// Kernel 3: flash attention with async double-buffered K/V tiles in LDS.
// Grid (T/128, B*H); 8 waves x 16 queries; Bc=32 keys per step.
// ---------------------------------------------------------------------------
__global__ __launch_bounds__(256) void k_attn(const bf16* __restrict__ Qr,
                                              const bf16* __restrict__ Kr,
                                              const bf16* __restrict__ Vt,
                                              bf16* __restrict__ Ob) {
  __shared__ __align__(16) bf16 sK[2][32 * 128];    // [j][d]   8 KB each
  __shared__ __align__(16) bf16 sV[2][128 * 32];    // [d][t]   8 KB each
  __shared__ __align__(16) bf16 plds[8][16 * 32];   // per-wave P bounce
  const int tid  = threadIdx.x;
  const int lane = tid & 31;
  const int wv   = tid >> 5;
  const int hi   = lane >> 4;
  const int ln   = lane & 15;
  const int bh = blockIdx.y;
  const int b  = bh >> 3;
  const int h  = bh & 7;
  const int q0 = blockIdx.x * 128 + wv * 16;
  const int myEnd = (q0 + 47) >> 5;
  const int kbMax = blockIdx.x * 4 + 4;

  const bf16* kbase = Kr + (size_t)bh * T_ * D_;
  const bf16* vbase = Vt + (size_t)bh * D_ * T_;

  auto stageKV = [&](int kb, int bsel) {
#pragma unroll
    for (int i = 0; i < 2; ++i) {                   // K: 8 KB contiguous
      int c = i * 256 + tid;
      async_b128(kbase + (size_t)kb * 32 * 128 + c * 8, &sK[bsel][c * 8]);
    }
#pragma unroll
    for (int i = 0; i < 2; ++i) {                   // V: 128 rows x 64 B
      int c = i * 256 + tid;
      int row = c >> 2, off = (c & 3) * 8;
      async_b128(vbase + (size_t)row * T_ + kb * 32 + off, &sV[bsel][row * 32 + off]);
    }
  };

  v16bf qf[4];
  const bf16* qrow = Qr + ((size_t)bh * T_ + q0 + ln) * D_;
#pragma unroll
  for (int kc = 0; kc < 4; ++kc) {
    const bf16* qp = qrow + kc * 32 + hi * 8;
    qf[kc] = load_frag(qp, qp + 16);
  }

  v8f oacc[8] = {};
  float m[8], l[8];
#pragma unroll
  for (int r = 0; r < 8; ++r) { m[r] = -3.0e38f; l[r] = 0.0f; }
  bf16* pl = &plds[wv][0];

  stageKV(0, 0);
  for (int kb = 0; kb < kbMax; ++kb) {
    int cur = kb & 1;
    if (kb + 1 < kbMax) { stageKV(kb + 1, cur ^ 1); wait_async<4>(); }
    else                { wait_async<0>(); }
    __syncthreads();
    if (kb < myEnd) {
      // ---- S = Q K^T (16x32), K-dim 128: batch 4 loads then 4 WMMAs ----
      v8f s0 = {}, s1 = {};
      v16bf kf[4];
#pragma unroll
      for (int kc = 0; kc < 4; ++kc) {
        const bf16* kp = &sK[cur][ln * 128 + kc * 32 + hi * 8];
        kf[kc] = load_frag(kp, kp + 16);
      }
#pragma unroll
      for (int kc = 0; kc < 4; ++kc) s0 = wmma_bf16(qf[kc], kf[kc], s0);
#pragma unroll
      for (int kc = 0; kc < 4; ++kc) {
        const bf16* kp = &sK[cur][(16 + ln) * 128 + kc * 32 + hi * 8];
        kf[kc] = load_frag(kp, kp + 16);
      }
#pragma unroll
      for (int kc = 0; kc < 4; ++kc) s1 = wmma_bf16(qf[kc], kf[kc], s1);
      // ---- online softmax ----
      const int j0 = kb * 32 + ln;
      const int j1 = j0 + 16;
#pragma unroll
      for (int r = 0; r < 8; ++r) {
        int qi = q0 + r + hi * 8;
        float e0 = s0[r], e1 = s1[r];
        if (j0 > qi) e0 = -1.0e30f;
        if (j1 > qi) e1 = -1.0e30f;
        float mx = fmaxf(e0, e1);
        mx = fmaxf(mx, __shfl_xor(mx, 1, 32));
        mx = fmaxf(mx, __shfl_xor(mx, 2, 32));
        mx = fmaxf(mx, __shfl_xor(mx, 4, 32));
        mx = fmaxf(mx, __shfl_xor(mx, 8, 32));
        float mnew = fmaxf(m[r], mx);
        float alpha = __expf(m[r] - mnew);
        float p0 = __expf(e0 - mnew);
        float p1 = __expf(e1 - mnew);
        float rs = p0 + p1;
        rs += __shfl_xor(rs, 1, 32);
        rs += __shfl_xor(rs, 2, 32);
        rs += __shfl_xor(rs, 4, 32);
        rs += __shfl_xor(rs, 8, 32);
        l[r] = l[r] * alpha + rs;
        m[r] = mnew;
#pragma unroll
        for (int ct = 0; ct < 8; ++ct) oacc[ct][r] *= alpha;
        int i = r + hi * 8;
        pl[i * 32 + ln]      = (bf16)p0;
        pl[i * 32 + ln + 16] = (bf16)p1;
      }
      asm volatile("s_wait_dscnt 0" ::: "memory");
      const bf16* pp = pl + ln * 32 + hi * 8;
      v16bf pf = load_frag(pp, pp + 16);
      // ---- O += P V: two groups of (4 loads, 4 WMMAs) ----
#pragma unroll
      for (int g = 0; g < 2; ++g) {
        v16bf vf[4];
#pragma unroll
        for (int u = 0; u < 4; ++u) {
          const bf16* vp = &sV[cur][((g * 4 + u) * 16 + ln) * 32 + hi * 8];
          vf[u] = load_frag(vp, vp + 16);
        }
#pragma unroll
        for (int u = 0; u < 4; ++u)
          oacc[g * 4 + u] = wmma_bf16(pf, vf[u], oacc[g * 4 + u]);
      }
    }
    __syncthreads();
  }

#pragma unroll
  for (int r = 0; r < 8; ++r) {
    float inv = 1.0f / l[r];
    int t = q0 + r + hi * 8;
    size_t rowoff = ((size_t)b * T_ + t) * 1024 + h * 128;
#pragma unroll
    for (int ct = 0; ct < 8; ++ct) {
      Ob[rowoff + ct * 16 + ln] = (bf16)(oacc[ct][r] * inv);
    }
  }
}

// ---------------------------------------------------------------------------
// Kernel 4: output projection with async LDS staging.
// out[4096,1024] = Ob[4096,1024] @ wo[1024,1024]; 64x128 tiles per block.
// ---------------------------------------------------------------------------
__global__ __launch_bounds__(256) void k_outproj(const bf16* __restrict__ Ob,
                                                 const bf16* __restrict__ wot,
                                                 float* __restrict__ out) {
  __shared__ __align__(16) bf16 sW[2][128 * 32];   // 8 KB each
  __shared__ __align__(16) bf16 sA[2][64 * 32];    // 4 KB each
  const int tid  = threadIdx.x;
  const int lane = tid & 31;
  const int wv   = tid >> 5;
  const int hi   = lane >> 4;
  const int ln   = lane & 15;
  const int rt   = wv & 3;
  const int half = wv >> 2;
  const int rowblk = blockIdx.x * 64;
  const int colblk = blockIdx.y * 128;
  const bf16* arows = Ob + (size_t)rowblk * 1024;
  const bf16* wcols = wot + (size_t)colblk * 1024;

  auto stage = [&](int k0, int bsel) {
#pragma unroll
    for (int i = 0; i < 2; ++i) {                  // W: 512 chunks
      int c = i * 256 + tid;
      int row = c >> 2, off = (c & 3) * 8;
      async_b128(wcols + (size_t)row * 1024 + k0 + off, &sW[bsel][row * 32 + off]);
    }
    {                                              // A: 256 chunks
      int row = tid >> 2, off = (tid & 3) * 8;
      async_b128(arows + (size_t)row * 1024 + k0 + off, &sA[bsel][row * 32 + off]);
    }
  };

  v8f acc[4] = {};
  stage(0, 0);
  for (int kk = 0; kk < 32; ++kk) {
    int cur = kk & 1;
    if (kk + 1 < 32) { stage((kk + 1) * 32, cur ^ 1); wait_async<3>(); }
    else             { wait_async<0>(); }
    __syncthreads();
    const bf16* ap = &sA[cur][(rt * 16 + ln) * 32 + hi * 8];
    v16bf af = load_frag(ap, ap + 16);
    v16bf bfr[4];
#pragma unroll
    for (int j = 0; j < 4; ++j) {
      const bf16* bp = &sW[cur][((half * 4 + j) * 16 + ln) * 32 + hi * 8];
      bfr[j] = load_frag(bp, bp + 16);
    }
#pragma unroll
    for (int j = 0; j < 4; ++j) acc[j] = wmma_bf16(af, bfr[j], acc[j]);
    __syncthreads();
  }
#pragma unroll
  for (int j = 0; j < 4; ++j) {
    int col = colblk + (half * 4 + j) * 16 + ln;
#pragma unroll
    for (int r = 0; r < 8; ++r) {
      int row = rowblk + rt * 16 + r + hi * 8;
      out[(size_t)row * 1024 + col] = acc[j][r];
    }
  }
}

// ---------------------------------------------------------------------------
extern "C" void kernel_launch(void* const* d_in, const int* in_sizes, int n_in,
                              void* d_out, int out_size, void* d_ws, size_t ws_size,
                              hipStream_t stream) {
  const float* x  = (const float*)d_in[0];
  const float* wq = (const float*)d_in[1];
  const float* wk = (const float*)d_in[2];
  const float* wv = (const float*)d_in[3];
  const float* wo = (const float*)d_in[4];
  float* out = (float*)d_out;

  char* ws = (char*)d_ws;
  bf16* xb  = (bf16*)(ws);                   //  8 MB  [B*T, M]
  bf16* wt  = (bf16*)(ws + 8388608);         //  6 MB  [H][384][1024]
  bf16* wot = (bf16*)(ws + 14680064);        //  2 MB  [1024][1024]
  bf16* Qr  = (bf16*)(ws + 16777216);        //  8 MB  [B,H,T,D]
  bf16* Kr  = (bf16*)(ws + 25165824);        //  8 MB  [B,H,T,D]
  bf16* Vt  = (bf16*)(ws + 33554432);        //  8 MB  [B,H,D,T]
  bf16* Ob  = (bf16*)(ws + 41943040);        //  8 MB  [B,T,H*D]  (48 MB total)

  k_convert<<<32768, 256, 0, stream>>>(x, wq, wk, wv, wo, xb, wt, wot);

  dim3 g1(64, 8, 2);
  k_qkv<<<g1, 256, 0, stream>>>(xb, wt, Qr, Kr, Vt);

  k_normrope<<<8192, 256, 0, stream>>>(Qr, Kr);

  dim3 g3(16, 16);
  k_attn<<<g3, 256, 0, stream>>>(Qr, Kr, Vt, Ob);

  dim3 g4(64, 8);
  k_outproj<<<g4, 256, 0, stream>>>(Ob, wot, out);
}
